// Attention_21174188769984
// MI455X (gfx1250) — matrix operand order repaired and checked by reading, not compiled, not run on hardware
//
#include <hip/hip_runtime.h>

// ---------------------------------------------------------------------------
// Problem: x [N=8192, D=4096] fp32
//   A = x^T x [D,D] ; B = softmax(A, axis=0) ; C = x @ B [N,D] fp32
// Compute-bound (~0.55 TFLOP vs ~100 MB traffic) -> bf16 WMMA, fp32 accum.
// ---------------------------------------------------------------------------
#define NROWS 8192
#define DDIM  4096

typedef __attribute__((ext_vector_type(4)))  unsigned int v4u;
typedef __attribute__((ext_vector_type(4)))  float        v4f;
typedef __attribute__((ext_vector_type(8)))  float        v8f;
typedef __attribute__((ext_vector_type(16))) __bf16       v16bf;

union FragU { v4u u[2]; v16bf v; };          // 32 B: one WMMA A/B operand
union Pack8 { v4u u; unsigned short s[8]; }; // 16 B: 8 bf16 lanes
union PackF { v4f f; float x[4]; };

static __device__ __forceinline__ unsigned short f2bf(float f) {
    union { float f; unsigned u; } v; v.f = f;
    unsigned r = v.u + 0x7FFFu + ((v.u >> 16) & 1u);
    return (unsigned short)(r >> 16);
}
static __device__ __forceinline__ float bf2f(unsigned short b) {
    union { unsigned u; float f; } v; v.u = ((unsigned)b) << 16;
    return v.f;
}

// ---- CDNA5 async global->LDS (ASYNCcnt-tracked DMA, 16 B per lane) --------
static __device__ __forceinline__ void async_b128(void* lds, const void* g) {
    // Low 32 bits of a generic pointer to LDS are the wave-relative LDS
    // byte address (aperture truncation rule, ISA 10.2).
    unsigned off = (unsigned)(size_t)lds;
    asm volatile("global_load_async_to_lds_b128 %0, %1, off"
                 :: "v"(off), "v"(g) : "memory");
}
static __device__ __forceinline__ void wait_async0() {
    asm volatile("s_wait_asynccnt 0x0" ::: "memory");
}

// ---------------------------------------------------------------------------
// Kernel 1: fp32 -> bf16 (8 elems/thread)
// ---------------------------------------------------------------------------
__global__ __launch_bounds__(256) void k_cvt_bf16(const float* __restrict__ in,
                                                  unsigned short* __restrict__ out,
                                                  long long total) {
    long long base = ((long long)blockIdx.x * 256 + threadIdx.x) * 8;
    if (base + 8 > total) return;
    PackF a, b;
    a.f = *(const v4f*)(in + base);
    b.f = *(const v4f*)(in + base + 4);
    Pack8 o;
    o.s[0] = f2bf(a.x[0]); o.s[1] = f2bf(a.x[1]);
    o.s[2] = f2bf(a.x[2]); o.s[3] = f2bf(a.x[3]);
    o.s[4] = f2bf(b.x[0]); o.s[5] = f2bf(b.x[1]);
    o.s[6] = f2bf(b.x[2]); o.s[7] = f2bf(b.x[3]);
    *(v4u*)(out + base) = o.u;
}

// ---------------------------------------------------------------------------
// Double-buffered WMMA GEMM (bf16 in, fp32 accumulate).
//   TRANS_A:  Aop[m][k] = Ag[k*lda + m]  (x^T in GEMM1; register scatter)
//   !TRANS_A: Aop[m][k] = Ag[m*lda + k]  (GEMM2; async DMA straight to LDS)
//   Bop[k][n] = Bg[k*ldb + n]            (always transposed-scatter into LDS)
// Block tile 128x128, BK=32, 256 threads = 8 waves; wave -> 32x64 (2x4 tiles),
// 8 v_wmma per K-step; one barrier per K-step.
// ---------------------------------------------------------------------------
#define BM 128
#define BN 128
#define BK 32
#define LPAD 8
#define LROW (BK + LPAD)   // 40 bf16 -> 80 B row stride (keeps 16 B alignment)

template <bool TRANS_A, bool OUT_BF16>
__global__ __launch_bounds__(256) void wmma_gemm_bf16(
    const unsigned short* __restrict__ Ag,
    const unsigned short* __restrict__ Bg,
    void* __restrict__ Cg,
    int M, int Ncol, int K, int lda, int ldb, int ldc) {

    __shared__ __align__(16) unsigned short As[2][BM * LROW];
    __shared__ __align__(16) unsigned short Bs[2][BN * LROW];

    const int tid  = threadIdx.x;
    const int lane = tid & 31;
    const int wave = tid >> 5;
    const int wm   = (wave & 3) << 5;   // 0,32,64,96
    const int wn   = (wave >> 2) << 6;  // 0,64
    const int m0   = blockIdx.y * BM;
    const int n0   = blockIdx.x * BN;

    v8f acc[2][4] = {};
    v4u areg[2];
    v4u breg[2];

    // Issue global traffic for the tile at k0 into LDS buffer `buf`:
    //  - !TRANS_A A-tile: async DMA directly into LDS (no VGPR round trip)
    //  - TRANS_A  A-tile: batched global loads into registers
    //  - B-tile:          batched global loads into registers
    auto issue = [&](int k0, int buf) {
        #pragma unroll
        for (int i = 0; i < 2; ++i) {
            const int c = tid + i * 256;             // 512 chunks of 8 bf16
            if constexpr (TRANS_A) {
                const int k  = c >> 4;
                const int mc = c & 15;
                areg[i] = *(const v4u*)&Ag[(size_t)(k0 + k) * lda + m0 + mc * 8];
            } else {
                const int row = c >> 2;
                const int kc  = (c & 3) << 3;
                async_b128(&As[buf][row * LROW + kc],
                           &Ag[(size_t)(m0 + row) * lda + k0 + kc]);
            }
        }
        #pragma unroll
        for (int i = 0; i < 2; ++i) {
            const int c  = tid + i * 256;
            const int k  = c >> 4;
            const int nc = c & 15;
            breg[i] = *(const v4u*)&Bg[(size_t)(k0 + k) * ldb + n0 + nc * 8];
        }
    };

    // Commit register-staged data into LDS buffer `buf` (transposed scatter).
    auto commit = [&](int buf) {
        if constexpr (TRANS_A) {
            #pragma unroll
            for (int i = 0; i < 2; ++i) {
                const int c  = tid + i * 256;
                const int k  = c >> 4;
                const int mc = c & 15;
                Pack8 t; t.u = areg[i];
                #pragma unroll
                for (int e = 0; e < 8; ++e)
                    As[buf][(mc * 8 + e) * LROW + k] = t.s[e];
            }
        }
        #pragma unroll
        for (int i = 0; i < 2; ++i) {
            const int c  = tid + i * 256;
            const int k  = c >> 4;
            const int nc = c & 15;
            Pack8 t; t.u = breg[i];
            #pragma unroll
            for (int e = 0; e < 8; ++e)
                Bs[buf][(nc * 8 + e) * LROW + k] = t.s[e];
        }
    };

    const int nT = K / BK;
    issue(0, 0);
    commit(0);
    if constexpr (!TRANS_A) wait_async0();
    __syncthreads();

    const int h  = lane >> 4;
    const int lr = lane & 15;

    for (int t = 0; t < nT; ++t) {
        const int cur = t & 1;
        // Prefetch tile t+1 into the other buffer (async) / registers.
        if (t + 1 < nT) issue((t + 1) * BK, 1 - cur);

        // Fragments per ISA layouts:
        //  A 16x32: lane-half h reads K = h*8..h*8+7 and 16+h*8..16+h*8+7
        //  B 32x16: lane col = lane&15, K = h*8..h*8+7 and 16+h*8..16+h*8+7
        FragU afr[2];
        #pragma unroll
        for (int tm = 0; tm < 2; ++tm) {
            const int m = wm + tm * 16 + lr;
            afr[tm].u[0] = *(const v4u*)&As[cur][m * LROW + h * 8];
            afr[tm].u[1] = *(const v4u*)&As[cur][m * LROW + 16 + h * 8];
        }
        FragU bfr[4];
        #pragma unroll
        for (int tn = 0; tn < 4; ++tn) {
            const int n = wn + tn * 16 + lr;
            bfr[tn].u[0] = *(const v4u*)&Bs[cur][n * LROW + h * 8];
            bfr[tn].u[1] = *(const v4u*)&Bs[cur][n * LROW + 16 + h * 8];
        }

        #pragma unroll
        for (int tm = 0; tm < 2; ++tm)
            #pragma unroll
            for (int tn = 0; tn < 4; ++tn)
                acc[tm][tn] = __builtin_amdgcn_wmma_f32_16x16x32_bf16(
                    false, afr[tm].v, false, bfr[tn].v,
                    (short)0, acc[tm][tn], false, false);

        // Drain staged registers into the next buffer after compute, so the
        // s_wait_loadcnt lands behind the WMMAs.
        if (t + 1 < nT) commit(1 - cur);
        // Own async copies must be complete before the barrier publishes the
        // next buffer to other waves.
        if constexpr (!TRANS_A) wait_async0();
        __syncthreads();
    }

    // Epilogue: C/D layout — VGPR r: lanes 0-15 -> M=r, lanes 16-31 -> M=8+r.
    #pragma unroll
    for (int tm = 0; tm < 2; ++tm) {
        #pragma unroll
        for (int tn = 0; tn < 4; ++tn) {
            #pragma unroll
            for (int r = 0; r < 8; ++r) {
                const int gm = m0 + wm + tm * 16 + r + h * 8;
                const int gn = n0 + wn + tn * 16 + lr;
                const float val = acc[tm][tn][r];
                if (OUT_BF16)
                    ((unsigned short*)Cg)[(size_t)gm * ldc + gn] = f2bf(val);
                else
                    ((float*)Cg)[(size_t)gm * ldc + gn] = val;
            }
        }
    }
    (void)M; (void)Ncol;
}

// ---------------------------------------------------------------------------
// Kernel 3: column softmax over bf16 scores [D,D], in place.
// ---------------------------------------------------------------------------
__global__ __launch_bounds__(256) void k_softmax_col(unsigned short* __restrict__ S,
                                                     int D) {
    __shared__ float red[4][64];
    const int tx = threadIdx.x & 63;
    const int ty = threadIdx.x >> 6;
    const int c  = blockIdx.x * 64 + tx;

    float mx = -3.0e38f;
    for (int i = ty; i < D; i += 4)
        mx = fmaxf(mx, bf2f(S[(size_t)i * D + c]));
    red[ty][tx] = mx;
    __syncthreads();
    if (ty == 0)
        red[0][tx] = fmaxf(fmaxf(red[0][tx], red[1][tx]),
                           fmaxf(red[2][tx], red[3][tx]));
    __syncthreads();
    mx = red[0][tx];
    __syncthreads();

    float sum = 0.0f;
    for (int i = ty; i < D; i += 4) {
        const size_t idx = (size_t)i * D + c;
        const float e = __expf(bf2f(S[idx]) - mx);
        sum += e;
        S[idx] = f2bf(e);
    }
    red[ty][tx] = sum;
    __syncthreads();
    if (ty == 0)
        red[0][tx] = red[0][tx] + red[1][tx] + red[2][tx] + red[3][tx];
    __syncthreads();
    const float inv = 1.0f / red[0][tx];

    for (int i = ty; i < D; i += 4) {
        const size_t idx = (size_t)i * D + c;
        S[idx] = f2bf(bf2f(S[idx]) * inv);
    }
}

// ---------------------------------------------------------------------------
// Launcher
// ---------------------------------------------------------------------------
extern "C" void kernel_launch(void* const* d_in, const int* in_sizes, int n_in,
                              void* d_out, int out_size, void* d_ws, size_t ws_size,
                              hipStream_t stream) {
    (void)in_sizes; (void)n_in; (void)out_size; (void)ws_size;

    const float* x = (const float*)d_in[0];
    float* out = (float*)d_out;

    // ws layout: [ x_bf16 : N*D*2 B ][ scores/probs bf16 : D*D*2 B ]  (96 MB)
    unsigned short* xbf    = (unsigned short*)d_ws;
    unsigned short* scores = (unsigned short*)((char*)d_ws + (size_t)NROWS * DDIM * 2);

    {
        const long long total = (long long)NROWS * DDIM;
        k_cvt_bf16<<<(int)(total / (256 * 8)), 256, 0, stream>>>(x, xbf, total);
    }
    {
        dim3 grid(DDIM / BN, DDIM / BM);
        wmma_gemm_bf16<true, true><<<grid, 256, 0, stream>>>(
            xbf, xbf, scores, DDIM, DDIM, NROWS, DDIM, DDIM, DDIM);
    }
    k_softmax_col<<<DDIM / 64, 256, 0, stream>>>(scores, DDIM);
    {
        dim3 grid(DDIM / BN, NROWS / BM);
        wmma_gemm_bf16<false, false><<<grid, 256, 0, stream>>>(
            xbf, scores, out, NROWS, DDIM, DDIM, DDIM, DDIM, DDIM);
    }
}
